// DyDCNv3_1047972020323
// MI455X (gfx1250) — matrix operand order repaired and checked by reading, not compiled, not run on hardware
//
#include <hip/hip_runtime.h>

typedef __attribute__((ext_vector_type(16))) _Float16 v16h;
typedef __attribute__((ext_vector_type(8)))  _Float16 v8h;
typedef __attribute__((ext_vector_type(8)))  float    v8f;

#define B_    8
#define C_    256
#define H_    64
#define W_    64
#define HW_   4096
#define KPTS  9
#define KTOT  2304     // C_ * KPTS
#define NCHUNK 72      // KTOT / 32

// ---------------------------------------------------------------------------
// Pack weight fp32 [O=256][Cin=256][9] -> f16 WMMA B-fragments, layout
// [chunk(72)][ntile(16)][lane(32)][e(16)], chunk q = kk*8 + cc (K = kk*256+c).
// B-fragment K order used here, same as A (16x16x32): lane<16 -> K=e, lane>=16 -> K=16+e,
// N = lane%16 within the n-tile.
// ---------------------------------------------------------------------------
__global__ void pack_weight_kernel(const float* __restrict__ w,
                                   _Float16* __restrict__ bp) {
    int u    = blockIdx.x * blockDim.x + threadIdx.x;   // < 72*16*32*16
    int e    = u & 15;
    int lane = (u >> 4) & 31;
    int nt   = (u >> 9) & 15;
    int q    = u >> 13;
    int kk   = q >> 3;                                  // kernel point
    int cc   = q & 7;                                   // 32-channel chunk
    int kB   = ((lane >= 16) ? 16 : 0) + e;             // B-fragment K map
    int c    = cc * 32 + kB;
    int o    = nt * 16 + (lane & 15);
    bp[u] = (_Float16)w[(o * C_ + c) * KPTS + kk];
}

// ---------------------------------------------------------------------------
// Fused deformable-sample + GEMM. One WG = 64 consecutive positions (one
// image row: shared b,h) x all 256 output channels. 8 waves: 4 M-subtiles x
// 2 N-halves; each wave accumulates 8 (16x16) f32 tiles via wmma f16.
//
// Thread -> A-fragment element mapping: t = tid*8 + j, so
//   e = (tid&1)*8 + j, al = (tid>>1)&31, ms = tid>>6  =>
//   ml (position) and kk are CONSTANT per thread per kernel point, and the
//   thread's 8 f16 results are one contiguous 16-byte LDS store.
// A-fragment K map used here (matches B): K = e + (al>=16 ? 16 : 0)
//   => c = cc*32 + (al>=16 ? 16 : 0) + (tid&1)*8 + j  (j = 0..7 consecutive).
// ---------------------------------------------------------------------------
__global__ void __launch_bounds__(256)
dcn_gemm_kernel(const float* __restrict__ x,
                const float* __restrict__ off,
                const float* __restrict__ msk,
                const _Float16* __restrict__ bp,
                float* __restrict__ out) {
    __shared__ __align__(32) _Float16 As[2][4 * 32 * 16];   // double buffer

    const int tid    = threadIdx.x;
    const int lane   = tid & 31;
    const int wave   = tid >> 5;
    const int m_sub  = wave & 3;
    const int n_half = wave >> 2;
    const int m0     = blockIdx.x * 64;
    const int b      = m0 >> 12;          // / 4096
    const int h      = (m0 >> 6) & 63;

    // writer-side decomposition of t = tid*8
    const int ehalf = tid & 1;
    const int al    = (tid >> 1) & 31;
    const int ms    = tid >> 6;
    const int ml    = ms * 16 + (al & 15);              // position in tile
    const int cbase = ((al >= 16) ? 16 : 0) + ehalf * 8;

    v8f acc[8];
    {
        v8f z = {};
        #pragma unroll
        for (int i = 0; i < 8; ++i) acc[i] = z;
    }

    const float* xb = x + (size_t)b * C_ * HW_;
    const v16h* bfrag = (const v16h*)bp;
    const int pbase = (h << 6) + ml;                    // (h, w=ml) in plane

    for (int kk = 0; kk < KPTS; ++kk) {
        // ---- per-thread bilinear metadata for (position ml, point kk) ----
        int   idxr[4];
        float wgtr[4];
        {
            int ky = kk / 3, kx = kk - ky * 3;
            float dy = off[(((b * 18) + kk * 2    ) << 12) + pbase];
            float dx = off[(((b * 18) + kk * 2 + 1) << 12) + pbase];
            float mv = msk[(((b * KPTS) + kk   ) << 12) + pbase];
            float py = (float)(h - 1 + ky) + dy;
            float px = (float)(ml - 1 + kx) + dx;
            float y0f = floorf(py), x0f = floorf(px);
            float ly = py - y0f, lx = px - x0f;
            int y0 = (int)y0f, x0 = (int)x0f;
            #pragma unroll
            for (int j = 0; j < 4; ++j) {
                int yc = y0 + (j >> 1), xc = x0 + (j & 1);
                bool valid = (yc >= 0) && (yc < H_) && (xc >= 0) && (xc < W_);
                int yi = min(max(yc, 0), H_ - 1);
                int xi = min(max(xc, 0), W_ - 1);
                float wy = (j >> 1) ? ly : (1.0f - ly);
                float wx = (j & 1) ? lx : (1.0f - lx);
                idxr[j] = yi * W_ + xi;
                wgtr[j] = valid ? (wy * wx * mv) : 0.0f;
            }
        }

        for (int cc = 0; cc < 8; ++cc) {
            const int q   = kk * 8 + cc;
            const int buf = q & 1;
            // ---- build this thread's 8 contiguous A-fragment f16s ----
            const float* xc = xb + (size_t)(cc * 32 + cbase) * HW_;
            v8h av;
            #pragma unroll
            for (int j = 0; j < 8; ++j) {
                const float* xcj = xc + (size_t)j * HW_;
                float v = wgtr[0] * xcj[idxr[0]]
                        + wgtr[1] * xcj[idxr[1]]
                        + wgtr[2] * xcj[idxr[2]]
                        + wgtr[3] * xcj[idxr[3]];
                av[j] = (_Float16)v;
            }
            *(v8h*)&As[buf][tid * 8] = av;              // one ds_store_b128
            __syncthreads();

            // ---- WMMA: 8 n-tiles per wave ----
            v16h a = *(const v16h*)&As[buf][(m_sub * 32 + lane) * 16];
            #pragma unroll
            for (int i = 0; i < 8; ++i) {
                int nt = n_half * 8 + i;
                v16h bf = bfrag[(size_t)((q * 16 + nt) * 32) + lane];
                acc[i] = __builtin_amdgcn_wmma_f32_16x16x32_f16(
                    false, a, false, bf, (short)0, acc[i], false, false);
            }
            // no trailing barrier: buffer (q&1) is next overwritten at q+2,
            // and the barrier at q+1 orders that against this read.
        }
    }

    // ---- store: C/D layout -> VGPR r holds M=r (lanes 0-15), M=r+8 (16-31)
    const int rbase = (lane >= 16) ? 8 : 0;
    #pragma unroll
    for (int i = 0; i < 8; ++i) {
        int o = (n_half * 8 + i) * 16 + (lane & 15);
        float* op = out + (((size_t)(b * C_ + o)) << 12) + (h << 6)
                        + m_sub * 16 + rbase;
        float4 lo, hi;
        lo.x = acc[i][0]; lo.y = acc[i][1]; lo.z = acc[i][2]; lo.w = acc[i][3];
        hi.x = acc[i][4]; hi.y = acc[i][5]; hi.z = acc[i][6]; hi.w = acc[i][7];
        *(float4*)(op)     = lo;                        // global_store_b128
        *(float4*)(op + 4) = hi;                        // global_store_b128
    }
}

// ---------------------------------------------------------------------------
// Per-channel BN statistics over (B,H,W): one block per channel.
// Emits fused scale = inv*gamma, bias = beta - mu*scale.
// ---------------------------------------------------------------------------
__global__ void reduce_kernel(const float* __restrict__ out,
                              const float* __restrict__ gamma,
                              const float* __restrict__ beta,
                              float* __restrict__ scale,
                              float* __restrict__ bias) {
    __shared__ float ss[256], sq[256];
    const int o = blockIdx.x;
    const int tid = threadIdx.x;
    float s = 0.0f, s2 = 0.0f;
    for (int b = 0; b < B_; ++b) {
        const float* p = out + (((size_t)(b * C_ + o)) << 12);
        for (int i = tid; i < HW_; i += 256) {
            float v = p[i];
            s += v; s2 += v * v;
        }
    }
    ss[tid] = s; sq[tid] = s2;
    __syncthreads();
    for (int st = 128; st > 0; st >>= 1) {
        if (tid < st) { ss[tid] += ss[tid + st]; sq[tid] += sq[tid + st]; }
        __syncthreads();
    }
    if (tid == 0) {
        float n   = (float)(B_ * HW_);
        float mu  = ss[0] / n;
        float var = sq[0] / n - mu * mu;
        float inv = rsqrtf(var + 1e-5f);
        float sc  = inv * gamma[o];
        scale[o] = sc;
        bias[o]  = beta[o] - mu * sc;
    }
}

__global__ void norm_kernel(float* __restrict__ out,
                            const float* __restrict__ scale,
                            const float* __restrict__ bias) {
    int idx = blockIdx.x * 256 + threadIdx.x;     // < B_*C_*HW_
    int o = (idx >> 12) & 255;
    out[idx] = out[idx] * scale[o] + bias[o];
}

// ---------------------------------------------------------------------------
extern "C" void kernel_launch(void* const* d_in, const int* in_sizes, int n_in,
                              void* d_out, int out_size, void* d_ws, size_t ws_size,
                              hipStream_t stream) {
    (void)in_sizes; (void)n_in; (void)out_size; (void)ws_size;
    const float* x     = (const float*)d_in[0];
    const float* off   = (const float*)d_in[1];
    const float* msk   = (const float*)d_in[2];
    const float* wgt   = (const float*)d_in[3];
    const float* gamma = (const float*)d_in[4];
    const float* beta  = (const float*)d_in[5];
    float* out = (float*)d_out;

    _Float16* bp  = (_Float16*)d_ws;                                  // 1.18 MB
    float* scale  = (float*)((char*)d_ws + (size_t)KTOT * C_ * 2);    // 256 f32
    float* bias   = scale + C_;

    pack_weight_kernel<<<(KTOT * C_) / 256, 256, 0, stream>>>(wgt, bp);
    dcn_gemm_kernel<<<(B_ * HW_) / 64, 256, 0, stream>>>(x, off, msk, bp, out);
    reduce_kernel<<<C_, 256, 0, stream>>>(out, gamma, beta, scale, bias);
    norm_kernel<<<(B_ * C_ * HW_) / 256, 256, 0, stream>>>(out, scale, bias);
}
